// IPAttnProcessor_13761075216748
// MI455X (gfx1250) — matrix operand order repaired and checked by reading, not compiled, not run on hardware
//
#include <hip/hip_runtime.h>
#include <hip/hip_bf16.h>

typedef __bf16 bf16_t;
typedef __attribute__((ext_vector_type(16))) __bf16 v16bf;
typedef __attribute__((ext_vector_type(8)))  __bf16 v8bf;
typedef __attribute__((ext_vector_type(8)))  float  v8f;

union FragA { v16bf v; v8bf h[2]; };

__device__ __forceinline__ v8f wmma_bf16(v16bf a, v16bf b, v8f c) {
  return __builtin_amdgcn_wmma_f32_16x16x32_bf16(false, a, false, b, (short)0, c, false, false);
}

#define TM 128
#define TN 128
#define TK 32

// C = A(MxK) * W(KxN), A is fp32 (or bf16 when A_BF16), W fp32; bf16 math on WMMA.
// out_mode 0: fp32 row-major MxN (+bias). out_mode 2: bf16 scatter [B,H,seq_len,160].
// rows_pb>0: A row m maps to batch b=m/rows_pb, row (m%rows_pb+row_off) with batch stride.
template<bool A_BF16>
__global__ __launch_bounds__(256) void gemm_kernel(
    const void* __restrict__ Aptr, const float* __restrict__ W,
    void* __restrict__ Outp, const float* __restrict__ bias,
    int M, int N, int K,
    int rows_pb, long a_batch_stride, int row_off,
    int out_mode, int seq_len)
{
  __shared__ bf16_t As[TM][TK];      // [m][k]
  __shared__ bf16_t BsT[TN][TK];     // [n][k]  (transposed for contiguous B-fragments)

  const int tid  = threadIdx.x;
  const int lane = tid & 31;
  const int wave = tid >> 5;
  const int wm   = wave >> 2;        // 0..1  -> 64 rows each
  const int wn   = wave & 3;         // 0..3  -> 32 cols each
  const int m0   = blockIdx.x * TM;
  const int n0   = blockIdx.y * TN;
  const int half = lane >> 4;
  const int l16  = lane & 15;

  v8f acc[4][2];
#pragma unroll
  for (int f = 0; f < 4; ++f)
#pragma unroll
    for (int g = 0; g < 2; ++g)
#pragma unroll
      for (int e = 0; e < 8; ++e) acc[f][g][e] = 0.0f;

  for (int k0 = 0; k0 < K; k0 += TK) {
    __syncthreads();
    if constexpr (A_BF16) {
      const bf16_t* A = (const bf16_t*)Aptr;
#pragma unroll
      for (int i = 0; i < 2; ++i) {
        int idx = (i * 256 + tid) * 8;          // bf16 elements (128x32 tile)
        int r = idx >> 5, kk = idx & 31;
        int grow = m0 + r;
        v8bf val;
#pragma unroll
        for (int e = 0; e < 8; ++e) val[e] = (bf16_t)0.0f;
        if (grow < M) val = *(const v8bf*)(A + (long)grow * K + k0 + kk);
        *(v8bf*)(&As[r][kk]) = val;
      }
    } else {
      const float* A = (const float*)Aptr;
#pragma unroll
      for (int i = 0; i < 4; ++i) {
        int idx = (i * 256 + tid) * 4;          // fp32 elements
        int r = idx >> 5, kk = idx & 31;
        int grow = m0 + r;
        float4 fv{0.f, 0.f, 0.f, 0.f};
        if (grow < M) {
          long arow;
          if (rows_pb > 0) {
            int b = grow / rows_pb;
            int rr = grow - b * rows_pb + row_off;
            arow = (long)b * a_batch_stride + (long)rr * K;
          } else {
            arow = (long)grow * K;
          }
          fv = *(const float4*)(A + arow + k0 + kk);
        }
        bf16_t* dst = &As[r][kk];
        dst[0] = (bf16_t)fv.x; dst[1] = (bf16_t)fv.y;
        dst[2] = (bf16_t)fv.z; dst[3] = (bf16_t)fv.w;
      }
    }
    // W tile (32 x 128 fp32), transpose into BsT as bf16
#pragma unroll
    for (int i = 0; i < 4; ++i) {
      int idx = i * 256 + tid;                  // float4 groups: 32 rows x 32 groups
      int kk = idx >> 5, ng = idx & 31;
      int n = ng * 4;
      float4 fv = *(const float4*)(W + (long)(k0 + kk) * N + n0 + n);
      BsT[n + 0][kk] = (bf16_t)fv.x;
      BsT[n + 1][kk] = (bf16_t)fv.y;
      BsT[n + 2][kk] = (bf16_t)fv.z;
      BsT[n + 3][kk] = (bf16_t)fv.w;
    }
    __syncthreads();

    FragA a[4], b[2];
#pragma unroll
    for (int f = 0; f < 4; ++f) {
      const bf16_t* p = &As[wm * 64 + f * 16 + l16][half * 16];
      a[f].h[0] = *(const v8bf*)p;
      a[f].h[1] = *(const v8bf*)(p + 8);
    }
#pragma unroll
    for (int g = 0; g < 2; ++g) {
      const bf16_t* p = &BsT[wn * 32 + g * 16 + l16][half * 16];
      b[g].h[0] = *(const v8bf*)p;
      b[g].h[1] = *(const v8bf*)(p + 8);
    }
#pragma unroll
    for (int f = 0; f < 4; ++f)
#pragma unroll
      for (int g = 0; g < 2; ++g)
        acc[f][g] = wmma_bf16(a[f].v, b[g].v, acc[f][g]);
  }

#pragma unroll
  for (int f = 0; f < 4; ++f)
#pragma unroll
    for (int g = 0; g < 2; ++g)
#pragma unroll
      for (int v = 0; v < 8; ++v) {
        int ml = wm * 64 + f * 16 + v + half * 8;
        int nl = wn * 32 + g * 16 + l16;
        int m = m0 + ml, n = n0 + nl;
        if (m >= M) continue;
        float val = acc[f][g][v];
        if (out_mode == 0) {
          ((float*)Outp)[(long)m * N + n] = val + bias[n];
        } else {
          int bb = m / seq_len; int s = m - bb * seq_len;
          int h  = n / 160;     int d = n - h * 160;
          ((bf16_t*)Outp)[(((long)(bb * 8 + h)) * seq_len + s) * 160 + d] = (bf16_t)val;
        }
      }
}

// Fused dual-softmax attention: 96 padded keys = 77 text (rows 0..76) + 4 ip (rows 80..83).
__global__ __launch_bounds__(256) void attn_kernel(
    const bf16_t* __restrict__ Q,                          // [B,H,S,160]
    const bf16_t* __restrict__ Kt, const bf16_t* __restrict__ Vt,   // [B,H,77,160]
    const bf16_t* __restrict__ Kip, const bf16_t* __restrict__ Vip, // [B,H,4,160]
    bf16_t* __restrict__ Out, int S)                       // [B,S,1280]
{
  __shared__ bf16_t Ks[96][160];     // [key][d]
  __shared__ bf16_t VsT[160][96];    // [d][key]  transposed for B-fragments
  __shared__ bf16_t Ps[8][16][96];   // per-wave probabilities

  const int tid = threadIdx.x, lane = tid & 31, wave = tid >> 5;
  const int bh = blockIdx.x;
  const long kvT = (long)bh * 77 * 160;
  const long kvI = (long)bh * 4 * 160;

  for (int idx = tid; idx < 96 * 160; idx += 256) {
    int r = idx / 160, d = idx - r * 160;
    bf16_t kv = (bf16_t)0.0f, vv = (bf16_t)0.0f;
    if (r < 77)                 { kv = Kt[kvT + (long)r * 160 + d];       vv = Vt[kvT + (long)r * 160 + d]; }
    else if (r >= 80 && r < 84) { kv = Kip[kvI + (long)(r - 80) * 160 + d]; vv = Vip[kvI + (long)(r - 80) * 160 + d]; }
    Ks[r][d]  = kv;
    VsT[d][r] = vv;
  }
  __syncthreads();

  const int half = lane >> 4, l16 = lane & 15;
  const int qrow = blockIdx.y * 128 + wave * 16 + l16;
  const long qbase = ((long)bh * S + qrow) * 160;

  FragA qa[5];
#pragma unroll
  for (int c = 0; c < 5; ++c) {
    const bf16_t* p = Q + qbase + c * 32 + half * 16;
    qa[c].h[0] = *(const v8bf*)p;
    qa[c].h[1] = *(const v8bf*)(p + 8);
  }

  v8f sc[6];
#pragma unroll
  for (int t = 0; t < 6; ++t) {
#pragma unroll
    for (int e = 0; e < 8; ++e) sc[t][e] = 0.0f;
#pragma unroll
    for (int c = 0; c < 5; ++c) {
      FragA kb;
      const bf16_t* p = &Ks[t * 16 + l16][c * 32 + half * 16];
      kb.h[0] = *(const v8bf*)p;
      kb.h[1] = *(const v8bf*)(p + 8);
      sc[t] = wmma_bf16(qa[c].v, kb.v, sc[t]);
    }
  }

  const float scale = 0.07905694150420949f;  // 1/sqrt(160)

  // text softmax over tiles 0..4 (cols < 77)
  {
    float pt[5][8], rmax[8], rsum[8];
#pragma unroll
    for (int e = 0; e < 8; ++e) rmax[e] = -3.0e38f;
#pragma unroll
    for (int t = 0; t < 5; ++t) {
      bool valid = (t * 16 + l16) < 77;
#pragma unroll
      for (int e = 0; e < 8; ++e) {
        float v = valid ? sc[t][e] * scale : -3.0e38f;
        pt[t][e] = v;
        rmax[e] = fmaxf(rmax[e], v);
      }
    }
#pragma unroll
    for (int off = 8; off >= 1; off >>= 1)
#pragma unroll
      for (int e = 0; e < 8; ++e) rmax[e] = fmaxf(rmax[e], __shfl_xor(rmax[e], off, 32));
#pragma unroll
    for (int e = 0; e < 8; ++e) rsum[e] = 0.0f;
#pragma unroll
    for (int t = 0; t < 5; ++t) {
      bool valid = (t * 16 + l16) < 77;
#pragma unroll
      for (int e = 0; e < 8; ++e) {
        float v = valid ? __expf(pt[t][e] - rmax[e]) : 0.0f;
        pt[t][e] = v;
        rsum[e] += v;
      }
    }
#pragma unroll
    for (int off = 8; off >= 1; off >>= 1)
#pragma unroll
      for (int e = 0; e < 8; ++e) rsum[e] += __shfl_xor(rsum[e], off, 32);
#pragma unroll
    for (int t = 0; t < 5; ++t)
#pragma unroll
      for (int e = 0; e < 8; ++e)
        Ps[wave][e + half * 8][t * 16 + l16] = (bf16_t)(pt[t][e] / rsum[e]);
  }
  // ip softmax over tile 5 (local cols < 4)
  {
    bool valid = l16 < 4;
    float pv[8], rmax[8], rsum[8];
#pragma unroll
    for (int e = 0; e < 8; ++e) { pv[e] = valid ? sc[5][e] * scale : -3.0e38f; rmax[e] = pv[e]; }
#pragma unroll
    for (int off = 8; off >= 1; off >>= 1)
#pragma unroll
      for (int e = 0; e < 8; ++e) rmax[e] = fmaxf(rmax[e], __shfl_xor(rmax[e], off, 32));
#pragma unroll
    for (int e = 0; e < 8; ++e) { pv[e] = valid ? __expf(pv[e] - rmax[e]) : 0.0f; rsum[e] = pv[e]; }
#pragma unroll
    for (int off = 8; off >= 1; off >>= 1)
#pragma unroll
      for (int e = 0; e < 8; ++e) rsum[e] += __shfl_xor(rsum[e], off, 32);
#pragma unroll
    for (int e = 0; e < 8; ++e)
      Ps[wave][e + half * 8][80 + l16] = (bf16_t)(pv[e] / rsum[e]);
  }
  __syncthreads();

  // P (16x96) @ Vall (96x160); text+ip summed implicitly (IP_SCALE == 1)
  FragA pa[3];
#pragma unroll
  for (int c = 0; c < 3; ++c) {
    const bf16_t* p = &Ps[wave][l16][c * 32 + half * 16];
    pa[c].h[0] = *(const v8bf*)p;
    pa[c].h[1] = *(const v8bf*)(p + 8);
  }
  v8f oacc[10];
#pragma unroll
  for (int dt = 0; dt < 10; ++dt) {
#pragma unroll
    for (int e = 0; e < 8; ++e) oacc[dt][e] = 0.0f;
#pragma unroll
    for (int c = 0; c < 3; ++c) {
      FragA vb;
      const bf16_t* p = &VsT[dt * 16 + l16][c * 32 + half * 16];
      vb.h[0] = *(const v8bf*)p;
      vb.h[1] = *(const v8bf*)(p + 8);
      oacc[dt] = wmma_bf16(pa[c].v, vb.v, oacc[dt]);
    }
  }

  const int b = bh >> 3, h = bh & 7;
  const int srow0 = blockIdx.y * 128 + wave * 16;
#pragma unroll
  for (int dt = 0; dt < 10; ++dt)
#pragma unroll
    for (int e = 0; e < 8; ++e) {
      int s = srow0 + e + half * 8;
      int d = dt * 16 + l16;
      Out[((long)b * S + s) * 1280 + h * 160 + d] = (bf16_t)oacc[dt][e];
    }
}

extern "C" void kernel_launch(void* const* d_in, const int* in_sizes, int n_in,
                              void* d_out, int out_size, void* d_ws, size_t ws_size,
                              hipStream_t stream) {
  const float* hidden = (const float*)d_in[0];
  const float* ehs    = (const float*)d_in[1];
  const float* w_q    = (const float*)d_in[2];
  const float* w_k    = (const float*)d_in[3];
  const float* w_v    = (const float*)d_in[4];
  const float* w_kip  = (const float*)d_in[5];
  const float* w_vip  = (const float*)d_in[6];
  const float* w_out  = (const float*)d_in[7];
  const float* b_out  = (const float*)d_in[8];
  float* out = (float*)d_out;

  const int Bn = 8, S = 4096, HS = 1280, LT = 77, NIP = 4;
  char* ws = (char*)d_ws;
  size_t off = 0;
  bf16_t* q_bf    = (bf16_t*)(ws + off); off += (size_t)Bn * 8 * S * 160 * 2;
  bf16_t* attn_bf = (bf16_t*)(ws + off); off += (size_t)Bn * S * HS * 2;
  bf16_t* kt_bf   = (bf16_t*)(ws + off); off += (size_t)Bn * 8 * LT * 160 * 2;
  bf16_t* vt_bf   = (bf16_t*)(ws + off); off += (size_t)Bn * 8 * LT * 160 * 2;
  bf16_t* kip_bf  = (bf16_t*)(ws + off); off += (size_t)Bn * 8 * NIP * 160 * 2;
  bf16_t* vip_bf  = (bf16_t*)(ws + off); off += (size_t)Bn * 8 * NIP * 160 * 2;

  dim3 blk(256);
  const long ehs_bs = (long)(LT + NIP) * 2048;

  // Q = hidden @ w_q  -> bf16 [B,H,S,160]
  gemm_kernel<false><<<dim3(256, 10), blk, 0, stream>>>(
      hidden, w_q, q_bf, b_out, 32768, 1280, 1280, 0, 0L, 0, 2, 4096);
  // K,V (text) = ehs[:, :77] @ w_k / w_v -> bf16 [B,H,77,160]
  gemm_kernel<false><<<dim3(5, 10), blk, 0, stream>>>(
      ehs, w_k, kt_bf, b_out, 616, 1280, 2048, 77, ehs_bs, 0, 2, 77);
  gemm_kernel<false><<<dim3(5, 10), blk, 0, stream>>>(
      ehs, w_v, vt_bf, b_out, 616, 1280, 2048, 77, ehs_bs, 0, 2, 77);
  // K,V (ip) = ehs[:, 77:81] @ w_k_ip / w_v_ip -> bf16 [B,H,4,160]
  gemm_kernel<false><<<dim3(1, 10), blk, 0, stream>>>(
      ehs, w_kip, kip_bf, b_out, 32, 1280, 2048, 4, ehs_bs, 77, 2, 4);
  gemm_kernel<false><<<dim3(1, 10), blk, 0, stream>>>(
      ehs, w_vip, vip_bf, b_out, 32, 1280, 2048, 4, ehs_bs, 77, 2, 4);
  // fused text+ip attention -> bf16 [B,S,1280]
  attn_kernel<<<dim3(64, 32), blk, 0, stream>>>(
      q_bf, kt_bf, vt_bf, kip_bf, vip_bf, attn_bf, 4096);
  // out = attn @ w_out + b_out -> fp32
  gemm_kernel<true><<<dim3(256, 10), blk, 0, stream>>>(
      attn_bf, w_out, out, b_out, 32768, 1280, 1280, 0, 0L, 0, 0, 0);

  (void)in_sizes; (void)n_in; (void)out_size; (void)ws_size;
}
